// LSTM_5703716569723
// MI455X (gfx1250) — matrix operand order repaired and checked by reading, not compiled, not run on hardware
//
#include <hip/hip_runtime.h>
#include <hip/hip_bf16.h>

// ---------------------------------------------------------------------------
// CDNA5 (gfx1250) AWD-LSTM forward: bf16 WMMA GEMMs + fused recurrent cell.
// ---------------------------------------------------------------------------

typedef __bf16 bf16_t;
typedef __attribute__((ext_vector_type(16))) bf16_t v16bf;
typedef __attribute__((ext_vector_type(8)))  float  v8f;
typedef __attribute__((ext_vector_type(8)))  unsigned short ushort8;

union Frag16 {
    v16bf   bf;
    ushort8 u8[2];
};

__device__ __forceinline__ unsigned short f2bf(float f) {
    unsigned int u = __builtin_bit_cast(unsigned int, f);
    u += 0x7fffu + ((u >> 16) & 1u);   // round-to-nearest-even
    return (unsigned short)(u >> 16);
}

__device__ __forceinline__ float sigf(float x) {
    return 1.0f / (1.0f + __expf(-x));
}

// A fragment: 16x32 bf16, row-major source [M, lda], rows m0..m0+15, cols k0..k0+31.
// ISA layout: lane<16 holds row=lane, K = {k0..k0+7} ∪ {k0+16..k0+23};
//             lane>=16 holds row=lane-16, K = {k0+8..k0+15} ∪ {k0+24..k0+31}.
__device__ __forceinline__ v16bf load_a_frag(const unsigned short* __restrict__ A,
                                             int lda, int m0, int k0, int lane) {
    int r  = lane & 15;
    int hi = lane >> 4;
    const unsigned short* p = A + (long)(m0 + r) * lda + k0 + hi * 8;
    Frag16 f;
    f.u8[0] = *(const ushort8*)(p);
    f.u8[1] = *(const ushort8*)(p + 16);
    return f.bf;
}

// B fragment: 32x16 bf16 from [K, ldb] row-major (i.e. weights pre-transposed).
// col = n0 + lane%16; K = {k0+8h..+7} ∪ {k0+16+8h..+7}, h = lane/16 (matches A).
__device__ __forceinline__ v16bf load_b_frag(const unsigned short* __restrict__ B,
                                             int ldb, int k0, int n0, int lane) {
    const unsigned short* p = B + (long)(k0 + (lane >> 4) * 8) * ldb + n0 + (lane & 15);
    Frag16 f;
    for (int j = 0; j < 2; ++j)
        for (int s = 0; s < 8; ++s)
            f.u8[j][s] = p[(long)(16 * j + s) * ldb];
    return f.bf;
}

__device__ __forceinline__ v8f wmma_bf16(v16bf a, v16bf b, v8f c) {
    return __builtin_amdgcn_wmma_f32_16x16x32_bf16(false, a, false, b, (short)0, c,
                                                   false, false);
}

// ---------------------------------------------------------------------------
// Generic bf16 GEMM: C[M,N] (f32) = A[M,K]bf16 * B[K,N]bf16 (+bias[N]).
// 128 threads = 4 waves; block tile 32x64; wave tile 16x32 (two 16x16 WMMAs).
// ---------------------------------------------------------------------------
__global__ void gemm_bf16_kernel(const unsigned short* __restrict__ A, int lda,
                                 const unsigned short* __restrict__ B, int ldb,
                                 float* __restrict__ C, long ldc,
                                 const float* __restrict__ bias,
                                 int K, int Nreal)
{
    const int lane = threadIdx.x & 31;
    const int w    = threadIdx.x >> 5;             // 0..3
    const int m0   = blockIdx.y * 32 + (w & 1) * 16;
    const int n0   = blockIdx.x * 64 + (w >> 1) * 32;

    v8f acc0 = {};
    v8f acc1 = {};
    for (int k = 0; k < K; k += 32) {
        v16bf a  = load_a_frag(A, lda, m0, k, lane);
        v16bf b0 = load_b_frag(B, ldb, k, n0,      lane);
        v16bf b1 = load_b_frag(B, ldb, k, n0 + 16, lane);
        acc0 = wmma_bf16(a, b0, acc0);
        acc1 = wmma_bf16(a, b1, acc1);
    }

    const int hi8 = (lane >> 4) * 8;
    const int nn  = lane & 15;
#pragma unroll
    for (int r = 0; r < 8; ++r) {
        int m  = m0 + r + hi8;
        int n  = n0 + nn;
        int n2 = n + 16;
        if (n  < Nreal) C[(long)m * ldc + n]  = acc0[r] + (bias ? bias[n]  : 0.0f);
        if (n2 < Nreal) C[(long)m * ldc + n2] = acc1[r] + (bias ? bias[n2] : 0.0f);
    }
}

// ---------------------------------------------------------------------------
// One LSTM timestep: g = xp[t] + h_prev * whhT, then cell nonlinearity.
// 256 threads = 8 waves. blockIdx.x picks a 16-wide hidden-unit strip; each
// wave owns (batch-half, gate) and WMMAs over K. Gates meet in LDS.
// h is double-buffered across launches (hprev read-only here).
// ---------------------------------------------------------------------------
__global__ void lstm_step_kernel(const unsigned short* __restrict__ hprev, // [32,1152] bf16
                                 const unsigned short* __restrict__ whhT,  // [Kp,Gp]  bf16
                                 const float* __restrict__ xp,             // [S*32,Gp] f32
                                 float* __restrict__ cst,                  // [32,1152] f32
                                 unsigned short* __restrict__ hnext,       // [32,1152] bf16
                                 unsigned short* __restrict__ xout,        // [S*32,1152] bf16
                                 int t, int Kp, int Hp, int Gp)
{
    __shared__ float g_lds[4 * 32 * 16];   // [gate][batch][k] = 8 KB

    const int lane = threadIdx.x & 31;
    const int w    = threadIdx.x >> 5;     // 0..7
    const int m0   = (w & 1) * 16;         // batch half
    const int ga   = w >> 1;               // gate 0..3 (i,f,g,o)
    const int kt   = blockIdx.x;           // hidden strip
    const int n0   = ga * Hp + kt * 16;

    v8f acc = {};
    for (int k = 0; k < Kp; k += 32) {
        v16bf a = load_a_frag(hprev, 1152, m0, k, lane);
        v16bf b = load_b_frag(whhT, Gp, k, n0, lane);
        acc = wmma_bf16(a, b, acc);
    }

    const int hi8 = (lane >> 4) * 8;
    const int nn  = lane & 15;
    const float* xpt = xp + (long)(t * 32) * Gp;
#pragma unroll
    for (int r = 0; r < 8; ++r) {
        int b = m0 + r + hi8;                       // batch index 0..31
        float g = acc[r] + xpt[(long)b * Gp + n0 + nn];
        g_lds[(ga * 32 + b) * 16 + nn] = g;
    }
    __syncthreads();

    for (int idx = threadIdx.x; idx < 512; idx += 256) {
        int b  = idx >> 4;
        int kk = idx & 15;
        int kc = kt * 16 + kk;
        float gi = g_lds[(0 * 32 + b) * 16 + kk];
        float gf = g_lds[(1 * 32 + b) * 16 + kk];
        float gg = g_lds[(2 * 32 + b) * 16 + kk];
        float go = g_lds[(3 * 32 + b) * 16 + kk];
        float cp = cst[b * 1152 + kc];
        float cn = sigf(gf) * cp + sigf(gi) * tanhf(gg);
        float h  = sigf(go) * tanhf(cn);
        cst[b * 1152 + kc] = cn;
        unsigned short hb = f2bf(h);
        hnext[b * 1152 + kc] = hb;
        xout[((long)(t * 32 + b)) * 1152 + kc] = hb;
    }
}

// ---------------------------------------------------------------------------
// Weight transpose + pad + (optional) mask + f32->bf16.
// src: [ngates*H, K] f32 ; dst: [Kp, ngates*Hp] bf16, zero-padded.
// ---------------------------------------------------------------------------
__global__ void transpose_pad_kernel(unsigned short* __restrict__ dst,
                                     const float* __restrict__ src,
                                     const float* __restrict__ mask,
                                     int K, int Kp, int H, int Hp, int ngates)
{
    const long N     = (long)ngates * Hp;
    const long total = (long)Kp * N;
    for (long i = (long)blockIdx.x * blockDim.x + threadIdx.x; i < total;
         i += (long)gridDim.x * blockDim.x) {
        long k = i / N;
        long n = i % N;
        int  g = (int)(n / Hp);
        int  r = (int)(n % Hp);
        float v = 0.0f;
        if (k < K && r < H) {
            long si = ((long)g * H + r) * K + k;
            v = src[si];
            if (mask) v *= mask[si];
        }
        dst[i] = f2bf(v);
    }
}

__global__ void bias_combine_kernel(float* __restrict__ dst,
                                    const float* __restrict__ bih,
                                    const float* __restrict__ bhh,
                                    int H, int Hp)
{
    int i = blockIdx.x * blockDim.x + threadIdx.x;
    if (i >= 4 * Hp) return;
    int g = i / Hp, r = i % Hp;
    dst[i] = (r < H) ? (bih[g * H + r] + bhh[g * H + r]) : 0.0f;
}

__global__ void embed_kernel(unsigned short* __restrict__ dst,   // [4096,1152]
                             const int* __restrict__ tokens,     // [S*B]
                             const float* __restrict__ emb)      // [V,400]
{
    long i = (long)blockIdx.x * blockDim.x + threadIdx.x;        // over 4096*416
    if (i >= 4096L * 416) return;
    int sb = (int)(i / 416);
    int e  = (int)(i % 416);
    float v = (e < 400) ? emb[(long)tokens[sb] * 400 + e] : 0.0f;
    dst[(long)sb * 1152 + e] = f2bf(v);
}

__global__ void zero_state_kernel(unsigned short* __restrict__ hs0,
                                  unsigned short* __restrict__ hs1,
                                  float* __restrict__ cst)
{
    int i = blockIdx.x * blockDim.x + threadIdx.x;
    if (i < 32 * 1152) { hs0[i] = 0; hs1[i] = 0; cst[i] = 0.0f; }
}

// ---------------------------------------------------------------------------
extern "C" void kernel_launch(void* const* d_in, const int* in_sizes, int n_in,
                              void* d_out, int out_size, void* d_ws, size_t ws_size,
                              hipStream_t stream)
{
    (void)in_sizes; (void)n_in; (void)out_size; (void)ws_size;

    const int SB = 4096;                 // S*B
    const int V  = 33278, Vp = 33280;

    const int*   tokens = (const int*)d_in[0];
    const float* emb    = (const float*)d_in[1];
    const float* fcb    = (const float*)d_in[2];
    const float* wih[3] = {(const float*)d_in[3],  (const float*)d_in[7],  (const float*)d_in[11]};
    const float* whh[3] = {(const float*)d_in[4],  (const float*)d_in[8],  (const float*)d_in[12]};
    const float* bih[3] = {(const float*)d_in[5],  (const float*)d_in[9],  (const float*)d_in[13]};
    const float* bhh[3] = {(const float*)d_in[6],  (const float*)d_in[10], (const float*)d_in[14]};
    const float* msk[3] = {(const float*)d_in[15], (const float*)d_in[16], (const float*)d_in[17]};

    char* ws = (char*)d_ws;
    size_t off = 0;
    auto alloc = [&](size_t bytes) -> void* {
        void* p = ws + off;
        off += (bytes + 255) & ~(size_t)255;
        return p;
    };
    unsigned short* bufA  = (unsigned short*)alloc((size_t)SB * 1152 * 2);
    unsigned short* bufB  = (unsigned short*)alloc((size_t)SB * 1152 * 2);
    float*          xp    = (float*)         alloc((size_t)SB * 4608 * 4);
    unsigned short* wihT  = (unsigned short*)alloc((size_t)1152 * 4608 * 2);
    unsigned short* whhT  = (unsigned short*)alloc((size_t)1152 * 4608 * 2);
    unsigned short* wembT = (unsigned short*)alloc((size_t)416 * Vp * 2);
    float*          biasb = (float*)         alloc((size_t)4608 * 4);
    unsigned short* hs0   = (unsigned short*)alloc((size_t)32 * 1152 * 2);
    unsigned short* hs1   = (unsigned short*)alloc((size_t)32 * 1152 * 2);
    float*          cst   = (float*)         alloc((size_t)32 * 1152 * 4);

    // Per-layer geometry (K padded to x32, hidden/gate padded to x16 per gate).
    const int Kin [3] = {400, 1150, 1150};
    const int Kpin[3] = {416, 1152, 1152};
    const int Hh  [3] = {1150, 1150, 400};
    const int Hpp [3] = {1152, 1152, 416};
    const int Gp  [3] = {4608, 4608, 1664};
    unsigned short* xin [3] = {bufA, bufB, bufA};
    unsigned short* xout[3] = {bufB, bufA, bufB};

    // Embedding gather -> bufA (bf16, K-padded with zeros).
    {
        long total = 4096L * 416;
        embed_kernel<<<(int)((total + 255) / 256), 256, 0, stream>>>(bufA, tokens, emb);
    }

    for (int l = 0; l < 3; ++l) {
        // w_ih^T (padded, bf16)
        {
            long total = (long)Kpin[l] * Gp[l];
            transpose_pad_kernel<<<(int)((total + 255) / 256), 256, 0, stream>>>(
                wihT, wih[l], nullptr, Kin[l], Kpin[l], Hh[l], Hpp[l], 4);
        }
        // (w_hh * mask)^T (padded, bf16)
        {
            long total = (long)Hpp[l] * Gp[l];
            transpose_pad_kernel<<<(int)((total + 255) / 256), 256, 0, stream>>>(
                whhT, whh[l], msk[l], Hh[l], Hpp[l], Hh[l], Hpp[l], 4);
        }
        // b_ih + b_hh (padded)
        bias_combine_kernel<<<(4 * Hpp[l] + 255) / 256, 256, 0, stream>>>(
            biasb, bih[l], bhh[l], Hh[l], Hpp[l]);

        // xp = x @ w_ih^T + (b_ih + b_hh)   — one big WMMA GEMM over all steps
        gemm_bf16_kernel<<<dim3(Gp[l] / 64, SB / 32), 128, 0, stream>>>(
            xin[l], 1152, wihT, Gp[l], xp, (long)Gp[l], biasb, Kpin[l], Gp[l]);

        // zero h (double-buffered) and c
        zero_state_kernel<<<(32 * 1152 + 255) / 256, 256, 0, stream>>>(hs0, hs1, cst);

        // sequential recurrence
        for (int t = 0; t < 128; ++t) {
            unsigned short* cur = (t & 1) ? hs1 : hs0;
            unsigned short* nxt = (t & 1) ? hs0 : hs1;
            lstm_step_kernel<<<Hpp[l] / 16, 256, 0, stream>>>(
                cur, whhT, xp, cst, nxt, xout[l], t, Hpp[l], Hpp[l], Gp[l]);
        }
    }

    // emb^T (padded, bf16) for tied projection
    {
        long total = 416L * Vp;
        transpose_pad_kernel<<<(int)((total + 255) / 256), 256, 0, stream>>>(
            wembT, emb, nullptr, 400, 416, V, Vp, 1);
    }

    // out[sb, v] = x2 @ emb^T + fc_bias  (f32 straight into d_out)
    gemm_bf16_kernel<<<dim3(Vp / 64, SB / 32), 128, 0, stream>>>(
        bufB, 1152, wembT, Vp, (float*)d_out, (long)V, fcb, 416, V);
}
